// KinematicsTrendAttention_41893111005326
// MI455X (gfx1250) — compile-verified, matrix operand order checked by reading, not run on hardware
//
#include <hip/hip_runtime.h>

typedef unsigned short u16;
typedef __bf16 bf16_t;
typedef bf16_t v16bf __attribute__((ext_vector_type(16)));
typedef float  v8f   __attribute__((ext_vector_type(8)));

union FragBF { v16bf v; uint4 q[2]; };

#define BB   4
#define NN   8
#define TT   512
#define DD   256
#define HH   8
#define DKK  32
#define BN   (BB * NN)          // 32
#define TOK  (BN * TT)          // 16384
#define D3   (3 * DD)           // 768

__device__ __forceinline__ u16 f2bf(float f) {
    union { float f; unsigned u; } v; v.f = f;
    unsigned r = v.u + 0x7FFFu + ((v.u >> 16) & 1u);   // round-to-nearest-even
    return (u16)(r >> 16);
}

// DPP row_ror-based reduction across the 16-lane row (VALU only, no LDS).
template <int CTRL>
__device__ __forceinline__ float dpp_rot(float x) {
    return __int_as_float(
        __builtin_amdgcn_update_dpp(0, __float_as_int(x), CTRL, 0xf, 0xf, true));
}
__device__ __forceinline__ float rowmax16(float x) {
    x = fmaxf(x, dpp_rot<0x121>(x));   // row_ror:1
    x = fmaxf(x, dpp_rot<0x122>(x));   // row_ror:2
    x = fmaxf(x, dpp_rot<0x124>(x));   // row_ror:4
    x = fmaxf(x, dpp_rot<0x128>(x));   // row_ror:8
    return x;
}

// ---------------------------------------------------------------------------
// Weight transpose + f32->bf16:  W (K x N) row-major  ->  WT (N x K) bf16
// ---------------------------------------------------------------------------
__global__ void transpose_to_bf16(const float* __restrict__ W, u16* __restrict__ WT,
                                  int K, int N) {
    int i = blockIdx.x * blockDim.x + threadIdx.x;
    if (i >= K * N) return;
    int k = i / N, n = i % N;
    WT[(size_t)n * K + k] = f2bf(W[i]);
}

__global__ void to_bf16(const float* __restrict__ x, u16* __restrict__ y, int n) {
    int i = blockIdx.x * blockDim.x + threadIdx.x;
    if (i < n) y[i] = f2bf(x[i]);
}

// ---------------------------------------------------------------------------
// Kinematic features: depthwise conv k=3 SAME (cross-correlation, as XLA)
// x (BN,T,D) f32  ->  out (BN,T,3D) bf16   [p | v | a]
// ---------------------------------------------------------------------------
__global__ void kinfeat_kernel(const float* __restrict__ x,
                               const float* __restrict__ w_pos,
                               const float* __restrict__ b_pos,
                               const float* __restrict__ w_vel,
                               const float* __restrict__ w_acc,
                               u16* __restrict__ out) {
    int i = blockIdx.x * blockDim.x + threadIdx.x;
    if (i >= TOK * DD) return;
    int d = i % DD;
    int token = i / DD;
    int t = token % TT;
    float xc = x[i];
    float xm = (t > 0)      ? x[i - DD] : 0.f;
    float xp = (t < TT - 1) ? x[i + DD] : 0.f;
    float p  = w_pos[d*3] * xm + w_pos[d*3+1] * xc + w_pos[d*3+2] * xp + b_pos[d];
    float vv = w_vel[d*3] * xm + w_vel[d*3+1] * xc + w_vel[d*3+2] * xp;
    float aa = w_acc[d*3] * xm + w_acc[d*3+1] * xc + w_acc[d*3+2] * xp;
    size_t ob = (size_t)token * D3;
    out[ob + d]          = f2bf(p);
    out[ob + DD + d]     = f2bf(vv);
    out[ob + 2 * DD + d] = f2bf(aa);
}

// ---------------------------------------------------------------------------
// Generic WMMA GEMM:  C(M x N) = A(M x K) * B + bias,   B supplied as BT (N x K)
// One wave per 16x64 output tile (A-fragment reused across 4 B-tiles).
// MODE 0: bf16 row-major out.  MODE 1: bf16 out in (bn,h,dk,t) transposed
// layout (for V).  MODE 2: f32 row-major out (final result).
// ---------------------------------------------------------------------------
template <int MODE>
__global__ __launch_bounds__(32)
void gemm_bf16(const u16* __restrict__ A, const u16* __restrict__ BT,
               const float* __restrict__ bias, void* __restrict__ Cp,
               int M, int N, int K) {
    const int lane = threadIdx.x;
    const int l16  = lane & 15;
    const int half = lane >> 4;
    const int m0 = blockIdx.x * 16;
    const int n0 = blockIdx.y * 64;

    const u16* arow  = A  + (size_t)(m0 + l16) * K + 8 * half;
    const u16* brow0 = BT + (size_t)(n0 + l16) * K + 8 * half;

    v8f z = {};
    v8f acc[4] = { z, z, z, z };

    #pragma unroll 2
    for (int k0 = 0; k0 < K; k0 += 32) {
        FragBF a;
        a.q[0] = *(const uint4*)(arow + k0);
        a.q[1] = *(const uint4*)(arow + k0 + 16);
        #pragma unroll
        for (int j = 0; j < 4; ++j) {
            FragBF b;
            const u16* bp = brow0 + (size_t)(16 * j) * K + k0;
            b.q[0] = *(const uint4*)(bp);
            b.q[1] = *(const uint4*)(bp + 16);
            acc[j] = __builtin_amdgcn_wmma_f32_16x16x32_bf16(
                false, a.v, false, b.v, (short)0, acc[j], false, false);
        }
    }

    #pragma unroll
    for (int j = 0; j < 4; ++j) {
        int c = n0 + 16 * j + l16;
        float bb = bias[c];
        #pragma unroll
        for (int r = 0; r < 8; ++r) {
            int row = m0 + r + 8 * half;               // token index
            float val = acc[j][r] + bb;
            if (MODE == 0) {
                ((u16*)Cp)[(size_t)row * N + c] = f2bf(val);
            } else if (MODE == 1) {
                int bn = row >> 9, tl = row & (TT - 1);
                int h = c >> 5, dk = c & (DKK - 1);
                ((u16*)Cp)[(((size_t)(bn * HH + h)) * DKK + dk) * TT + tl] = f2bf(val);
            } else {
                ((float*)Cp)[(size_t)row * N + c] = val;
            }
        }
    }
}

// ---------------------------------------------------------------------------
// Flash-style attention head tile. One wave handles 16 query rows of one
// (bn, h). Streams keys in chunks of 64 with online softmax.
// Row-sum of P is computed with an extra WMMA against an all-ones B matrix
// (lands in C-layout, matching the o/corr layout exactly); row-max uses a
// DPP row_ror rotate-reduce (pure VALU).
//   Q, K : (BN*T, D) bf16 row-major     VT : (BN, H, DK, T) bf16
//   ctx  : (BN*T, D) bf16 row-major
// ---------------------------------------------------------------------------
__global__ __launch_bounds__(32)
void attn_kernel(const u16* __restrict__ Q, const u16* __restrict__ Kb,
                 const u16* __restrict__ VT, const int* __restrict__ mask,
                 u16* __restrict__ ctx) {
    __shared__ __align__(16) u16 Pl[16 * 64];

    const int lane = threadIdx.x;
    const int l16  = lane & 15;
    const int half = lane >> 4;
    const int t0 = blockIdx.x * 16;
    const int h  = blockIdx.y;
    const int bn = blockIdx.z;

    const u16* Qh = Q  + (size_t)bn * TT * DD + h * DKK;
    const u16* Kh = Kb + (size_t)bn * TT * DD + h * DKK;
    const u16* Vh = VT + ((size_t)(bn * HH + h)) * DKK * TT;
    const int* mrow = mask + bn * TT;

    FragBF aq;                                   // Q fragment: reused all chunks
    {
        const u16* p = Qh + (size_t)(t0 + l16) * DD + 8 * half;
        aq.q[0] = *(const uint4*)(p);
        aq.q[1] = *(const uint4*)(p + 16);
    }
    FragBF ones;                                 // all-ones bf16 B fragment
    {
        uint4 q; q.x = q.y = q.z = q.w = 0x3F803F80u;
        ones.q[0] = q; ones.q[1] = q;
    }

    v8f o0 = {}, o1 = {};                        // ctx accumulators (dk 0-15 / 16-31)
    v8f lacc = {};                               // softmax denominators (C-layout)
    float m[8];
    #pragma unroll
    for (int r = 0; r < 8; ++r) m[r] = -3.0e38f;
    const float scale = 0.17677669529663687f;    // 1/sqrt(32)

    for (int k0 = 0; k0 < TT; k0 += 64) {
        // --- S(16x64) = Q * K^T  (four 16x16 WMMAs, K dim = DK = 32)
        v8f s[4];
        #pragma unroll
        for (int j = 0; j < 4; ++j) {
            FragBF bk;
            const u16* pk = Kh + (size_t)(k0 + 16 * j + l16) * DD + 8 * half;
            bk.q[0] = *(const uint4*)(pk);
            bk.q[1] = *(const uint4*)(pk + 16);
            v8f z = {};
            s[j] = __builtin_amdgcn_wmma_f32_16x16x32_bf16(
                false, aq.v, false, bk.v, (short)0, z, false, false);
        }
        int mk[4];
        #pragma unroll
        for (int j = 0; j < 4; ++j) mk[j] = mrow[k0 + 16 * j + l16];

        // --- online softmax: per-row max via DPP, exp, correction
        #pragma unroll
        for (int r = 0; r < 8; ++r) {
            float sv[4];
            float cm = -3.0e38f;
            #pragma unroll
            for (int j = 0; j < 4; ++j) {
                float v0 = s[j][r] * scale;
                if (mk[j]) v0 = -1e9f;
                sv[j] = v0;
                cm = fmaxf(cm, v0);
            }
            cm = rowmax16(cm);                    // max over the 16 columns
            float mn   = fmaxf(m[r], cm);
            float corr = __expf(m[r] - mn);
            m[r] = mn;
            o0[r]   *= corr;
            o1[r]   *= corr;
            lacc[r] *= corr;
            #pragma unroll
            for (int j = 0; j < 4; ++j) {
                float e = __expf(sv[j] - mn);
                Pl[(r + 8 * half) * 64 + 16 * j + l16] = f2bf(e);
            }
        }
        asm volatile("s_wait_dscnt 0" ::: "memory");

        // --- C-layout -> A-layout: P fragments (keys 0..31 / 32..63)
        FragBF ap0, ap1;
        const u16* pp = &Pl[l16 * 64 + 8 * half];
        ap0.q[0] = *(const uint4*)(pp);
        ap0.q[1] = *(const uint4*)(pp + 16);
        ap1.q[0] = *(const uint4*)(pp + 32);
        ap1.q[1] = *(const uint4*)(pp + 48);

        // --- softmax denominator: row-sum of P via ones-matrix WMMA
        lacc = __builtin_amdgcn_wmma_f32_16x16x32_bf16(
            false, ap0.v, false, ones.v, (short)0, lacc, false, false);
        lacc = __builtin_amdgcn_wmma_f32_16x16x32_bf16(
            false, ap1.v, false, ones.v, (short)0, lacc, false, false);

        // --- ctx += P(16x64) * V(64x32): B-frags from transposed V (dk-major)
        {
            const u16* pv0 = Vh + (size_t)l16 * TT + k0 + 8 * half;   // dk 0-15
            const u16* pv1 = pv0 + (size_t)16 * TT;                   // dk 16-31
            FragBF bv;
            bv.q[0] = *(const uint4*)(pv0);
            bv.q[1] = *(const uint4*)(pv0 + 16);
            o0 = __builtin_amdgcn_wmma_f32_16x16x32_bf16(
                false, ap0.v, false, bv.v, (short)0, o0, false, false);
            bv.q[0] = *(const uint4*)(pv0 + 32);
            bv.q[1] = *(const uint4*)(pv0 + 48);
            o0 = __builtin_amdgcn_wmma_f32_16x16x32_bf16(
                false, ap1.v, false, bv.v, (short)0, o0, false, false);
            bv.q[0] = *(const uint4*)(pv1);
            bv.q[1] = *(const uint4*)(pv1 + 16);
            o1 = __builtin_amdgcn_wmma_f32_16x16x32_bf16(
                false, ap0.v, false, bv.v, (short)0, o1, false, false);
            bv.q[0] = *(const uint4*)(pv1 + 32);
            bv.q[1] = *(const uint4*)(pv1 + 48);
            o1 = __builtin_amdgcn_wmma_f32_16x16x32_bf16(
                false, ap1.v, false, bv.v, (short)0, o1, false, false);
        }
        asm volatile("s_wait_dscnt 0" ::: "memory");   // P reads done before rewrite
    }

    // --- normalize + write ctx (bf16, natural (token, h*DK+dk) layout)
    #pragma unroll
    for (int r = 0; r < 8; ++r) {
        float inv = 1.0f / lacc[r];
        int row = t0 + r + 8 * half;
        u16* op = ctx + (size_t)(bn * TT + row) * DD + h * DKK;
        op[l16]      = f2bf(o0[r] * inv);
        op[16 + l16] = f2bf(o1[r] * inv);
    }
}

// ---------------------------------------------------------------------------
extern "C" void kernel_launch(void* const* d_in, const int* in_sizes, int n_in,
                              void* d_out, int out_size, void* d_ws, size_t ws_size,
                              hipStream_t stream) {
    const float* query = (const float*)d_in[0];
    const float* key_  = (const float*)d_in[1];
    const float* value = (const float*)d_in[2];
    const int*   mask  = (const int*)d_in[3];
    const float* w_pos = (const float*)d_in[4];
    const float* b_pos = (const float*)d_in[5];
    const float* w_vel = (const float*)d_in[6];
    const float* w_acc = (const float*)d_in[7];
    const float* Wq = (const float*)d_in[8];
    const float* bq = (const float*)d_in[9];
    const float* Wk = (const float*)d_in[10];
    const float* bk = (const float*)d_in[11];
    const float* Wv = (const float*)d_in[12];
    const float* bv = (const float*)d_in[13];
    const float* Wo = (const float*)d_in[14];
    const float* bo = (const float*)d_in[15];

    char* ws = (char*)d_ws;
    size_t off = 0;
    auto alloc = [&](size_t bytes) -> void* {
        void* p = ws + off;
        off += (bytes + 255) & ~(size_t)255;
        return p;
    };
    u16* qc  = (u16*)alloc((size_t)TOK * D3 * 2);   // kin feats of query
    u16* kc  = (u16*)alloc((size_t)TOK * D3 * 2);   // kin feats of key
    u16* WqT = (u16*)alloc((size_t)DD * D3 * 2);
    u16* WkT = (u16*)alloc((size_t)DD * D3 * 2);
    u16* WvT = (u16*)alloc((size_t)DD * DD * 2);
    u16* WoT = (u16*)alloc((size_t)DD * DD * 2);
    u16* vbf = (u16*)alloc((size_t)TOK * DD * 2);   // value in bf16
    u16* Qb  = (u16*)alloc((size_t)TOK * DD * 2);
    u16* Kb  = (u16*)alloc((size_t)TOK * DD * 2);
    u16* VT  = (u16*)alloc((size_t)TOK * DD * 2);   // (BN,H,DK,T)
    u16* ctx = (u16*)alloc((size_t)TOK * DD * 2);

    // 1. weights -> bf16 transposed (N x K)
    transpose_to_bf16<<<dim3((D3 * DD + 255) / 256), 256, 0, stream>>>(Wq, WqT, D3, DD);
    transpose_to_bf16<<<dim3((D3 * DD + 255) / 256), 256, 0, stream>>>(Wk, WkT, D3, DD);
    transpose_to_bf16<<<dim3((DD * DD + 255) / 256), 256, 0, stream>>>(Wv, WvT, DD, DD);
    transpose_to_bf16<<<dim3((DD * DD + 255) / 256), 256, 0, stream>>>(Wo, WoT, DD, DD);

    // 2. kinematic features (p|v|a) in bf16
    kinfeat_kernel<<<dim3((TOK * DD + 255) / 256), 256, 0, stream>>>(
        query, w_pos, b_pos, w_vel, w_acc, qc);
    kinfeat_kernel<<<dim3((TOK * DD + 255) / 256), 256, 0, stream>>>(
        key_, w_pos, b_pos, w_vel, w_acc, kc);
    to_bf16<<<dim3((TOK * DD + 255) / 256), 256, 0, stream>>>(value, vbf, TOK * DD);

    // 3. projections (WMMA GEMMs)
    gemm_bf16<0><<<dim3(TOK / 16, DD / 64), 32, 0, stream>>>(qc, WqT, bq, Qb, TOK, DD, D3);
    gemm_bf16<0><<<dim3(TOK / 16, DD / 64), 32, 0, stream>>>(kc, WkT, bk, Kb, TOK, DD, D3);
    gemm_bf16<1><<<dim3(TOK / 16, DD / 64), 32, 0, stream>>>(vbf, WvT, bv, VT, TOK, DD, DD);

    // 4. masked attention, flash-style per (bn, h, 16-row tile)
    attn_kernel<<<dim3(TT / 16, HH, BN), 32, 0, stream>>>(Qb, Kb, VT, mask, ctx);

    // 5. output projection -> f32 d_out
    gemm_bf16<2><<<dim3(TOK / 16, DD / 64), 32, 0, stream>>>(ctx, WoT, bo, d_out, TOK, DD, DD);
}
